// MultimodalFusionNet_36129264894305
// MI455X (gfx1250) — compile-verified, matrix operand order checked by reading, not run on hardware
//
#include <hip/hip_runtime.h>
#include <hip/hip_bf16.h>
#include <math.h>

// ---------------------------------------------------------------------------
// MI455X / gfx1250. Convs + GEMMs via v_wmma_f32_16x16x32_f16 (f16 in, f32 acc).
// Conv input halo tiles staged global->LDS with GLOBAL_LOAD_ASYNC_TO_LDS
// (ASYNCcnt); weights staged in LDS.
// ---------------------------------------------------------------------------

typedef __attribute__((ext_vector_type(16))) _Float16 v16h;
typedef __attribute__((ext_vector_type(8)))  float    v8f;

#define EPSV 1e-5f

// K mapping for 16-bit A/B fragments (ISA 7.12.2, 16-bit A 16x32 table):
//   half h of lane L covers K = (h&7) + ((h>>3)<<4) + 8*(L>=16)
__device__ __forceinline__ int kmap(int h, int hi) {
    return (h & 7) + ((h >> 3) << 4) + hi * 8;
}

// ---- async global->LDS 4-byte copy (falls back to sync copy) --------------
__device__ __forceinline__ void async_copy_b32(const void* g, void* l) {
#if __has_builtin(__builtin_amdgcn_global_load_async_to_lds_b32)
    __builtin_amdgcn_global_load_async_to_lds_b32(
        (__attribute__((address_space(1))) int*)const_cast<void*>(g),
        (__attribute__((address_space(3))) int*)l, 0, 0);
#else
    *(unsigned*)l = *(const unsigned*)g;
#endif
}

__device__ __forceinline__ void async_wait0() {
#if __has_builtin(__builtin_amdgcn_s_wait_asynccnt)
    __builtin_amdgcn_s_wait_asynccnt(0);
#else
    asm volatile("s_wait_asynccnt 0x0" ::: "memory");
#endif
}

template<int INF32> struct InT { typedef _Float16 T; };
template<>          struct InT<1> { typedef float T; };

// ---------------------------------------------------------------------------
// Implicit-GEMM 3x3 SAME conv, NCHW. Block = 128 thr (4 waves) covering one
// 8x8 pixel tile (64 positions) x 16 output channels. Input halo (CIN x 10 x 12,
// origin x0-2 so every b32 chunk is fully valid or fully padding) staged in LDS
// via async loads; weights staged f32->f16 in LDS. Each wave: one 16x16 WMMA
// tile, K = CIN*9 padded to 32.
// ---------------------------------------------------------------------------
template<int CIN, int COUT, int HWs, int INF32>
__global__ __launch_bounds__(128) void conv3x3_wmma(const void* __restrict__ xin,
                                                    const float* __restrict__ w,
                                                    _Float16* __restrict__ y) {
    constexpr int KC   = CIN * 9;
    constexpr int KPAD = (KC + 31) & ~31;
    constexpr int LROW = 10, LCOL = 12;
    typedef typename InT<INF32>::T IT;

    __shared__ _Float16 wlds[16 * KPAD];
    __shared__ IT       xlds[CIN * LROW * LCOL];

    const int tpr = HWs / 8;                       // tiles per row
    const int b   = blockIdx.x / (tpr * tpr);
    const int r2  = blockIdx.x % (tpr * tpr);
    const int y0  = (r2 / tpr) * 8, x0 = (r2 % tpr) * 8;
    const int coutBase = blockIdx.y * 16;
    const int HW2 = HWs * HWs;

    // zero halo (padding) + stage weights (f32 -> f16)
    for (int idx = threadIdx.x; idx < CIN * LROW * LCOL; idx += 128) xlds[idx] = (IT)0;
    for (int idx = threadIdx.x; idx < 16 * KPAD; idx += 128) {
        int m = idx / KPAD, k = idx % KPAD;
        wlds[idx] = (_Float16)((k < KC) ? w[(size_t)(coutBase + m) * KC + k] : 0.f);
    }
    __syncthreads();                                // zeros visible before async overwrite

    // stage input halo window: rows y0-1..y0+8, cols x0-2..x0+9
    if (INF32) {                                    // f32 elements (conv1, CIN=1)
        const float* gx = (const float*)xin;
        for (int idx = threadIdx.x; idx < CIN * LROW * LCOL; idx += 128) {
            int cin = idx / (LROW * LCOL), rr = (idx / LCOL) % LROW, cc = idx % LCOL;
            int yy = y0 + rr - 1, xx = x0 + cc - 2;
            if (yy >= 0 && yy < HWs && xx >= 0 && xx < HWs)
                async_copy_b32(gx + ((size_t)b * CIN + cin) * HW2 + (size_t)yy * HWs + xx,
                               (float*)xlds + idx);
        }
    } else {                                        // f16: b32 chunk = 2 halves, 6 chunks/row
        const _Float16* gx = (const _Float16*)xin;
        for (int idx = threadIdx.x; idx < CIN * LROW * 6; idx += 128) {
            int cin = idx / (LROW * 6), rr = (idx / 6) % LROW, ch = idx % 6;
            int yy = y0 + rr - 1, xx = x0 + ch * 2 - 2;
            if (yy >= 0 && yy < HWs && xx >= 0 && (xx + 1) < HWs)
                async_copy_b32(gx + ((size_t)b * CIN + cin) * HW2 + (size_t)yy * HWs + xx,
                               (_Float16*)xlds + cin * LROW * LCOL + rr * LCOL + ch * 2);
        }
    }
    async_wait0();
    __syncthreads();

    const unsigned lane = threadIdx.x & 31;
    const int wave = threadIdx.x >> 5;
    const int hi = (lane >> 4) & 1;
    const int m  = lane & 15;
    const int n  = wave * 16 + (lane & 15);        // 0..63 position in 8x8 tile
    const int ly = n >> 3, lx = n & 7;

    v8f acc = {};
    for (int kb = 0; kb < KPAD; kb += 32) {
        v16h a, bf;
#pragma unroll
        for (int h = 0; h < 16; ++h) {
            int k = kb + kmap(h, hi);
            a[h] = wlds[m * KPAD + k];
            float v = 0.f;
            if (k < KC) {
                int cin = k / 9, r9 = k % 9;
                int rr = r9 / 3, ss = r9 % 3;
                int li = cin * LROW * LCOL + (ly + rr) * LCOL + (lx + ss + 1);
                v = (float)xlds[li];
            }
            bf[h] = (_Float16)v;
        }
        acc = __builtin_amdgcn_wmma_f32_16x16x32_f16(false, a, false, bf,
                                                     (short)0, acc, false, false);
    }

    const int yy = y0 + ly, xx = x0 + lx;
#pragma unroll
    for (int vv = 0; vv < 8; ++vv) {
        int cout = coutBase + vv + 8 * hi;
        y[(((size_t)b * COUT + cout) * HWs + yy) * HWs + xx] = (_Float16)acc[vv];
    }
}

// ---------------------------------------------------------------------------
// Generic WMMA GEMM: C[M,N] = act(A[M,K] * W[N,K]^T + bias[N]), f32 in/out.
// ACT: 0=none 1=relu 2=silu.  One wave per 16x16 tile, 4 waves/block on N.
// ---------------------------------------------------------------------------
template<int ACT>
__global__ __launch_bounds__(128) void gemm_wmma_k(const float* __restrict__ A,
                                                   const float* __restrict__ W,
                                                   const float* __restrict__ bias,
                                                   float* __restrict__ C,
                                                   int M, int N, int K) {
    const unsigned lane = threadIdx.x & 31;
    const int wave  = threadIdx.x >> 5;
    const int ntile = blockIdx.y * 4 + wave;
    if (ntile * 16 >= N) return;                   // wave-uniform
    const int mtile = blockIdx.x;
    const int hi = (lane >> 4) & 1;
    const int rA = mtile * 16 + (lane & 15);
    const int rW = ntile * 16 + (lane & 15);
    const bool va = rA < M, vw = rW < N;

    v8f acc = {};
    const int Ksteps = (K + 31) >> 5;
    for (int ks = 0; ks < Ksteps; ++ks) {
        int kb = ks << 5;
        if (va && kb + 32 < K) __builtin_prefetch(&A[(size_t)rA * K + kb + 32], 0, 3);
        v16h a, bfr;
#pragma unroll
        for (int h = 0; h < 16; ++h) {
            int k = kb + kmap(h, hi);
            a[h]   = (_Float16)((va && k < K) ? A[(size_t)rA * K + k] : 0.f);
            bfr[h] = (_Float16)((vw && k < K) ? W[(size_t)rW * K + k] : 0.f);
        }
        acc = __builtin_amdgcn_wmma_f32_16x16x32_f16(false, a, false, bfr,
                                                     (short)0, acc, false, false);
    }
    const int col = ntile * 16 + (lane & 15);
#pragma unroll
    for (int vv = 0; vv < 8; ++vv) {
        int row = mtile * 16 + vv + 8 * hi;
        if (row < M && col < N) {
            float x = acc[vv] + (bias ? bias[col] : 0.f);
            if (ACT == 1) x = fmaxf(x, 0.f);
            else if (ACT == 2) x = x / (1.f + expf(-x));
            C[(size_t)row * N + col] = x;
        }
    }
}

// ---------------------------------------------------------------------------
// BatchNorm stats over (B, H, W) per channel, input f16.
// ---------------------------------------------------------------------------
__global__ __launch_bounds__(256) void bn_stats(const _Float16* __restrict__ x,
                                                float* __restrict__ mean,
                                                float* __restrict__ var,
                                                int C, int HW2) {
    const int c = blockIdx.x;
    float s = 0.f, sq = 0.f;
    for (int bb = 0; bb < 64; ++bb) {
        const _Float16* row = x + ((size_t)bb * C + c) * HW2;
        for (int sp = threadIdx.x; sp < HW2; sp += 256) {
            float v = (float)row[sp];
            s += v; sq += v * v;
        }
    }
    __shared__ float s1[256], s2[256];
    s1[threadIdx.x] = s; s2[threadIdx.x] = sq; __syncthreads();
    for (int off = 128; off > 0; off >>= 1) {
        if (threadIdx.x < off) { s1[threadIdx.x] += s1[threadIdx.x + off];
                                 s2[threadIdx.x] += s2[threadIdx.x + off]; }
        __syncthreads();
    }
    if (threadIdx.x == 0) {
        float cnt = 64.f * HW2;
        float m = s1[0] / cnt;
        mean[c] = m;
        var[c]  = s2[0] / cnt - m * m;
    }
}

// BN + ReLU + 2x2 maxpool, f16 -> f16
__global__ __launch_bounds__(256) void bn_relu_pool(const _Float16* __restrict__ x,
                                                    const float* __restrict__ mean,
                                                    const float* __restrict__ var,
                                                    const float* __restrict__ g,
                                                    const float* __restrict__ beta,
                                                    _Float16* __restrict__ y,
                                                    int C, int Hin) {
    const int Ho = Hin >> 1;
    const long total = 64L * C * Ho * Ho;
    long i = (long)blockIdx.x * 256 + threadIdx.x;
    if (i >= total) return;
    int xo = (int)(i % Ho); long t = i / Ho;
    int yo = (int)(t % Ho); t /= Ho;
    int c  = (int)(t % C);  int bb = (int)(t / C);
    float m = mean[c], iv = rsqrtf(var[c] + EPSV), gg = g[c], be = beta[c];
    const _Float16* base = x + (((size_t)bb * C + c) * Hin + yo * 2) * Hin + xo * 2;
    float v0 = fmaxf(((float)base[0]       - m) * iv * gg + be, 0.f);
    float v1 = fmaxf(((float)base[1]       - m) * iv * gg + be, 0.f);
    float v2 = fmaxf(((float)base[Hin]     - m) * iv * gg + be, 0.f);
    float v3 = fmaxf(((float)base[Hin + 1] - m) * iv * gg + be, 0.f);
    y[(((size_t)bb * C + c) * Ho + yo) * Ho + xo] = (_Float16)fmaxf(fmaxf(v0, v1), fmaxf(v2, v3));
}

// BN + ReLU + global average pool, f16 -> f32 feature (one block per (b,c))
__global__ __launch_bounds__(256) void bn_relu_gap(const _Float16* __restrict__ x,
                                                   const float* __restrict__ mean,
                                                   const float* __restrict__ var,
                                                   const float* __restrict__ g,
                                                   const float* __restrict__ beta,
                                                   float* __restrict__ feat,
                                                   int C, int HW2) {
    const int bc = blockIdx.x;
    const int c  = bc % C;
    float m = mean[c], iv = rsqrtf(var[c] + EPSV), gg = g[c], be = beta[c];
    float s = 0.f;
    const _Float16* row = x + (size_t)bc * HW2;
    for (int sp = threadIdx.x; sp < HW2; sp += 256)
        s += fmaxf(((float)row[sp] - m) * iv * gg + be, 0.f);
    __shared__ float sm[256];
    sm[threadIdx.x] = s; __syncthreads();
    for (int off = 128; off > 0; off >>= 1) {
        if (threadIdx.x < off) sm[threadIdx.x] += sm[threadIdx.x + off];
        __syncthreads();
    }
    if (threadIdx.x == 0) feat[bc] = sm[0] / (float)HW2;
}

// LayerNorm over width 64 (block=64), optional residual, strided in/out.
__global__ __launch_bounds__(64) void layernorm64(const float* __restrict__ x, int xs,
                                                  const float* __restrict__ res, int rs,
                                                  const float* __restrict__ g,
                                                  const float* __restrict__ b,
                                                  float* __restrict__ y, int ys) {
    const int r = blockIdx.x, t = threadIdx.x;
    float v = x[(size_t)r * xs + t] + (res ? res[(size_t)r * rs + t] : 0.f);
    __shared__ float s1[64], s2[64];
    s1[t] = v; s2[t] = v * v; __syncthreads();
    for (int off = 32; off > 0; off >>= 1) {
        if (t < off) { s1[t] += s1[t + off]; s2[t] += s2[t + off]; }
        __syncthreads();
    }
    float m  = s1[0] * (1.f / 64.f);
    float vr = s2[0] * (1.f / 64.f) - m * m;
    y[(size_t)r * ys + t] = (v - m) * rsqrtf(vr + EPSV) * g[t] + b[t];
}

// GAT layer (B blocks x 64 threads): tiny, scalar f32.
__global__ __launch_bounds__(64) void gat_kernel(const float* __restrict__ x,   // (B,3,64)
                                                 const float* __restrict__ W,   // (64,64)
                                                 const float* __restrict__ a,   // (32)
                                                 float* __restrict__ out) {     // (B,3,64)
    const int b = blockIdx.x, o = threadIdx.x;
    __shared__ float h[3][64], sei[3][4], sej[3][4];
    for (int n = 0; n < 3; ++n) {
        float s = 0.f;
        const float* xr = x + ((size_t)b * 3 + n) * 64;
        const float* wr = W + (size_t)o * 64;
        for (int i2 = 0; i2 < 64; ++i2) s += xr[i2] * wr[i2];
        h[n][o] = s;
    }
    __syncthreads();
    if (o < 12) {
        int n = o >> 2, hh = o & 3; float s = 0.f;
        for (int d = 0; d < 16; ++d) s += h[n][hh * 16 + d] * a[d];
        sei[n][hh] = s;
    } else if (o < 24) {
        int q = o - 12; int n = q >> 2, hh = q & 3; float s = 0.f;
        for (int d = 0; d < 16; ++d) s += h[n][hh * 16 + d] * a[16 + d];
        sej[n][hh] = s;
    }
    __syncthreads();
    const int hh = o >> 4;
    for (int i2 = 0; i2 < 3; ++i2) {
        float e0 = sei[i2][hh] + sej[0][hh];
        float e1 = sei[i2][hh] + sej[1][hh];
        float e2 = sei[i2][hh] + sej[2][hh];
        e0 = e0 >= 0.f ? e0 : 0.2f * e0;
        e1 = e1 >= 0.f ? e1 : 0.2f * e1;
        e2 = e2 >= 0.f ? e2 : 0.2f * e2;
        float mx = fmaxf(e0, fmaxf(e1, e2));
        float p0 = expf(e0 - mx), p1 = expf(e1 - mx), p2 = expf(e2 - mx);
        float inv = 1.f / (p0 + p1 + p2);
        float s = (p0 * h[0][o] + p1 * h[1][o] + p2 * h[2][o]) * inv;
        out[((size_t)b * 3 + i2) * 64 + o] = s >= 0.f ? s : expm1f(s);
    }
}

// Build denoiser input u = [z_t | cond | time-emb]  (B blocks x 224 threads)
__global__ __launch_bounds__(224) void diff_prep(const float* __restrict__ raw,   // (B,3,64)
                                                 const float* __restrict__ noise, // (3,B,64)
                                                 const int* __restrict__ td,      // (3,B)
                                                 float* __restrict__ U, int i) {
    const int b = blockIdx.x, t = threadIdx.x;
    const int tt = td[i * 64 + b];
    float ab = 1.f;
    for (int k = 0; k <= tt; ++k)
        ab *= 1.f - (1e-4f + (float)k * ((0.02f - 1e-4f) / 99.f));
    float* u = U + (size_t)b * 224;
    if (t < 64) {
        u[t] = sqrtf(ab)  * raw[(size_t)b * 192 + i * 64 + t]
             + sqrtf(1.f - ab) * noise[((size_t)i * 64 + b) * 64 + t];
    } else if (t < 192) {
        int c = t - 64; int e = c >> 6; int j = e + (e >= i ? 1 : 0);
        u[t] = raw[(size_t)b * 192 + j * 64 + (c & 63)];
    } else {
        int q = t - 192;
        int j = q & 15;
        float fr = expf(-logf(10000.f) * (float)j / 15.f);
        float arg = (float)tt * fr;
        u[t] = (q < 16) ? sinf(arg) : cosf(arg);
    }
}

__global__ void zero1(float* p) { if (threadIdx.x == 0 && blockIdx.x == 0) p[0] = 0.f; }

__global__ __launch_bounds__(256) void mse_acc(const float* __restrict__ p,
                                               const float* __restrict__ nz,
                                               float* __restrict__ acc, int n) {
    float s = 0.f;
    for (int i = blockIdx.x * 256 + threadIdx.x; i < n; i += gridDim.x * 256) {
        float d = p[i] - nz[i]; s += d * d;
    }
    __shared__ float sm[256];
    sm[threadIdx.x] = s; __syncthreads();
    for (int off = 128; off > 0; off >>= 1) {
        if (threadIdx.x < off) sm[threadIdx.x] += sm[threadIdx.x + off];
        __syncthreads();
    }
    if (threadIdx.x == 0) atomicAdd(acc, sm[0]);
}

__global__ __launch_bounds__(64) void l2norm_rows(const float* __restrict__ x,
                                                  float* __restrict__ y) {
    const int r = blockIdx.x, t = threadIdx.x;
    float v = x[(size_t)r * 64 + t];
    __shared__ float sm[64];
    sm[t] = v * v; __syncthreads();
    for (int off = 32; off > 0; off >>= 1) {
        if (t < off) sm[t] += sm[t + off];
        __syncthreads();
    }
    float nrm = fmaxf(sqrtf(sm[0]), 1e-12f);
    y[(size_t)r * 64 + t] = v / nrm;
}

// log-sum-exp over rows (y=0) / cols (y=1) of 192x192 sim (scaled 1/0.07)
__global__ __launch_bounds__(256) void lse192(const float* __restrict__ S,
                                              float* __restrict__ lr,
                                              float* __restrict__ lc,
                                              float* __restrict__ dg) {
    const int idx = blockIdx.x, isCol = blockIdx.y, t = threadIdx.x;
    float val = 0.f, v = -INFINITY;
    if (t < 192) {
        val = (isCol ? S[(size_t)t * 192 + idx] : S[(size_t)idx * 192 + t]) * (1.f / 0.07f);
        v = val;
    }
    __shared__ float sm[256];
    sm[t] = v; __syncthreads();
    for (int off = 128; off > 0; off >>= 1) {
        if (t < off) sm[t] = fmaxf(sm[t], sm[t + off]);
        __syncthreads();
    }
    float mx = sm[0]; __syncthreads();
    float e = (t < 192) ? expf(val - mx) : 0.f;
    sm[t] = e; __syncthreads();
    for (int off = 128; off > 0; off >>= 1) {
        if (t < off) sm[t] += sm[t + off];
        __syncthreads();
    }
    if (t == 0) (isCol ? lc : lr)[idx] = mx + logf(sm[0]);
    if (!isCol && t == idx) dg[idx] = val;
}

__global__ __launch_bounds__(256) void finalize(const float* __restrict__ acc,
                                                const float* __restrict__ lr,
                                                const float* __restrict__ lc,
                                                const float* __restrict__ dg,
                                                float* __restrict__ out) {
    const int t = threadIdx.x;
    float v = (t < 192) ? (lr[t] - dg[t]) + (lc[t] - dg[t]) : 0.f;
    __shared__ float sm[256];
    sm[t] = v; __syncthreads();
    for (int off = 128; off > 0; off >>= 1) {
        if (t < off) sm[t] += sm[t + off];
        __syncthreads();
    }
    if (t == 0) {
        out[193] = sm[0] / (2.f * 192.f);           // graph loss
        out[192] = acc[0] / (3.f * 64.f * 64.f);    // diff loss
    }
}

// ---------------------------------------------------------------------------
// Host side
// ---------------------------------------------------------------------------
extern "C" void kernel_launch(void* const* d_in, const int* in_sizes, int n_in,
                              void* d_out, int out_size, void* d_ws, size_t ws_size,
                              hipStream_t stream) {
    (void)in_sizes; (void)n_in; (void)out_size; (void)ws_size;

    // ---- input pytree leaves (jax sorted-key order) ----
    const float* img[3] = { (const float*)d_in[0], (const float*)d_in[1], (const float*)d_in[2] };
    const float* noise  = (const float*)d_in[3];
    const float* clf_b1 = (const float*)d_in[4];
    const float* clf_b2 = (const float*)d_in[5];
    const float* clf_b3 = (const float*)d_in[6];
    const float* clf_w1 = (const float*)d_in[7];
    const float* clf_w2 = (const float*)d_in[8];
    const float* clf_w3 = (const float*)d_in[9];
    auto den = [&](int i, int j) { return (const float*)d_in[10 + 6 * i + j]; };  // b1,b2,b3,w1,w2,w3
    auto enc = [&](int i, int j) { return (const float*)d_in[28 + 16 * i + j]; }; // bn1b..lng
    const float* gatW[2] = { (const float*)d_in[76], (const float*)d_in[78] };
    const float* gatA[2] = { (const float*)d_in[77], (const float*)d_in[79] };
    const float* n1b = (const float*)d_in[80];
    const float* n1g = (const float*)d_in[81];
    const float* n2b = (const float*)d_in[82];
    const float* n2g = (const float*)d_in[83];
    const int*   tdiff = (const int*)d_in[84];
    float* out = (float*)d_out;

    // ---- workspace layout ----
    char* ws = (char*)d_ws;
    constexpr size_t SZ_BUFA = 102760448ull * 2;   // B*32*224*224 f16
    constexpr size_t SZ_BUFB = 25690112ull * 2;    // B*32*112*112 f16
    constexpr size_t OFF_BUFA = 0;
    constexpr size_t OFF_BUFB = OFF_BUFA + SZ_BUFA;
    constexpr size_t OFF_MV   = OFF_BUFB + SZ_BUFB;
    constexpr size_t OFF_FEAT = OFF_MV + 1024;
    constexpr size_t OFF_FCO  = OFF_FEAT + 64 * 128 * 4;
    constexpr size_t OFF_RAW  = OFF_FCO + 64 * 64 * 4;
    constexpr size_t OFF_GOUT = OFF_RAW + 64 * 192 * 4;
    constexpr size_t OFF_H1   = OFF_GOUT + 64 * 192 * 4;
    constexpr size_t OFF_GF   = OFF_H1 + 64 * 192 * 4;
    constexpr size_t OFF_U    = OFF_GF + 64 * 192 * 4;
    constexpr size_t OFF_DH1  = OFF_U + 64 * 224 * 4;
    constexpr size_t OFF_DH2  = OFF_DH1 + 64 * 256 * 4;
    constexpr size_t OFF_PRED = OFF_DH2 + 64 * 256 * 4;
    constexpr size_t OFF_ZB   = OFF_PRED + 64 * 64 * 4;
    constexpr size_t OFF_ZA   = OFF_ZB + 192 * 64 * 4;
    constexpr size_t OFF_SIM  = OFF_ZA + 192 * 64 * 4;
    constexpr size_t OFF_LSER = OFF_SIM + 192 * 192 * 4;
    constexpr size_t OFF_LSEC = OFF_LSER + 192 * 4;
    constexpr size_t OFF_DIAG = OFF_LSEC + 192 * 4;
    constexpr size_t OFF_ACC  = OFF_DIAG + 192 * 4 + 64;
    constexpr size_t OFF_C1   = OFF_ACC + 256;
    constexpr size_t OFF_C2   = OFF_C1 + 64 * 256 * 4;

    _Float16* bufA = (_Float16*)(ws + OFF_BUFA);
    _Float16* bufB = (_Float16*)(ws + OFF_BUFB);
    float* mean = (float*)(ws + OFF_MV);
    float* var  = mean + 128;
    float* feat = (float*)(ws + OFF_FEAT);
    float* fco  = (float*)(ws + OFF_FCO);
    float* raw  = (float*)(ws + OFF_RAW);
    float* gout = (float*)(ws + OFF_GOUT);
    float* h1   = (float*)(ws + OFF_H1);
    float* gf   = (float*)(ws + OFF_GF);
    float* U    = (float*)(ws + OFF_U);
    float* dh1  = (float*)(ws + OFF_DH1);
    float* dh2  = (float*)(ws + OFF_DH2);
    float* pred = (float*)(ws + OFF_PRED);
    float* zb   = (float*)(ws + OFF_ZB);
    float* za   = (float*)(ws + OFF_ZA);
    float* Smat = (float*)(ws + OFF_SIM);
    float* lser = (float*)(ws + OFF_LSER);
    float* lsec = (float*)(ws + OFF_LSEC);
    float* diag = (float*)(ws + OFF_DIAG);
    float* acc  = (float*)(ws + OFF_ACC);
    float* c1   = (float*)(ws + OFF_C1);
    float* c2   = (float*)(ws + OFF_C2);

    // ---- encoders ----
    for (int i = 0; i < 3; ++i) {
        // conv1 1->32 @224  (bias omitted: BN cancels per-channel bias exactly)
        conv3x3_wmma<1, 32, 224, 1><<<dim3(64 * 28 * 28, 2), 128, 0, stream>>>(img[i], enc(i, 7), bufA);
        bn_stats<<<32, 256, 0, stream>>>(bufA, mean, var, 32, 224 * 224);
        bn_relu_pool<<<100352, 256, 0, stream>>>(bufA, mean, var, enc(i, 1), enc(i, 0), bufB, 32, 224);
        // conv2 32->64 @112
        conv3x3_wmma<32, 64, 112, 0><<<dim3(64 * 14 * 14, 4), 128, 0, stream>>>(bufB, enc(i, 9), bufA);
        bn_stats<<<64, 256, 0, stream>>>(bufA, mean, var, 64, 112 * 112);
        bn_relu_pool<<<50176, 256, 0, stream>>>(bufA, mean, var, enc(i, 3), enc(i, 2), bufB, 64, 112);
        // conv3 64->128 @56
        conv3x3_wmma<64, 128, 56, 0><<<dim3(64 * 7 * 7, 8), 128, 0, stream>>>(bufB, enc(i, 11), bufA);
        bn_stats<<<128, 256, 0, stream>>>(bufA, mean, var, 128, 56 * 56);
        bn_relu_gap<<<64 * 128, 256, 0, stream>>>(bufA, mean, var, enc(i, 5), enc(i, 4), feat, 128, 56 * 56);
        // fc (64x128 @ (64,128)^T) + LN -> raw[:, i]
        gemm_wmma_k<0><<<dim3(4, 1), 128, 0, stream>>>(feat, enc(i, 13), enc(i, 12), fco, 64, 64, 128);
        layernorm64<<<64, 64, 0, stream>>>(fco, 64, nullptr, 0, enc(i, 15), enc(i, 14), raw + i * 64, 192);
    }

    // ---- GAT stack ----
    gat_kernel<<<64, 64, 0, stream>>>(raw, gatW[0], gatA[0], gout);
    layernorm64<<<192, 64, 0, stream>>>(gout, 64, raw, 64, n1g, n1b, h1, 64);
    gat_kernel<<<64, 64, 0, stream>>>(h1, gatW[1], gatA[1], gout);
    layernorm64<<<192, 64, 0, stream>>>(gout, 64, h1, 64, n2g, n2b, gf, 64);

    // ---- diffusion denoisers ----
    zero1<<<1, 1, 0, stream>>>(acc);
    for (int i = 0; i < 3; ++i) {
        diff_prep<<<64, 224, 0, stream>>>(raw, noise, tdiff, U, i);
        gemm_wmma_k<2><<<dim3(4, 4), 128, 0, stream>>>(U,   den(i, 3), den(i, 0), dh1,  64, 256, 224);
        gemm_wmma_k<2><<<dim3(4, 4), 128, 0, stream>>>(dh1, den(i, 4), den(i, 1), dh2,  64, 256, 256);
        gemm_wmma_k<0><<<dim3(4, 1), 128, 0, stream>>>(dh2, den(i, 5), den(i, 2), pred, 64, 64, 256);
        mse_acc<<<16, 256, 0, stream>>>(pred, noise + (size_t)i * 64 * 64, acc, 64 * 64);
    }

    // ---- contrastive graph loss ----
    l2norm_rows<<<192, 64, 0, stream>>>(raw, zb);
    l2norm_rows<<<192, 64, 0, stream>>>(gf, za);
    gemm_wmma_k<0><<<dim3(12, 3), 128, 0, stream>>>(zb, za, nullptr, Smat, 192, 192, 64);
    lse192<<<dim3(192, 2), 256, 0, stream>>>(Smat, lser, lsec, diag);

    // ---- classifier -> logits ----
    gemm_wmma_k<1><<<dim3(4, 4), 128, 0, stream>>>(gf, clf_w1, clf_b1, c1, 64, 256, 192);
    gemm_wmma_k<1><<<dim3(4, 2), 128, 0, stream>>>(c1, clf_w2, clf_b2, c2, 64, 128, 256);
    gemm_wmma_k<0><<<dim3(4, 1), 128, 0, stream>>>(c2, clf_w3, clf_b3, out, 64, 3, 128);

    finalize<<<1, 256, 0, stream>>>(acc, lser, lsec, diag, out);
}
